// MultiHeadCrossAttention_76158360092790
// MI455X (gfx1250) — matrix operand order repaired and checked by reading, not compiled, not run on hardware
//
#include <hip/hip_runtime.h>
#include <hip/hip_bf16.h>

// Problem constants (match reference)
#define B_   16
#define SQ_  1024
#define SE_  1024
#define C_   256      // C_IN == C_ENC
#define DK_  64
#define DV_  64
#define H_   8
#define F_   512      // H * DV
#define BN_EPS    1e-5f
#define NEG_SLOPE 0.01f

typedef __attribute__((ext_vector_type(16))) __bf16 v16bf;
typedef __attribute__((ext_vector_type(8)))  __bf16 v8bf;
typedef __attribute__((ext_vector_type(4)))  __bf16 v4bf;
typedef __attribute__((ext_vector_type(8)))  float  v8f;

static __device__ __forceinline__ v8f vzero8() {
  v8f z = {0.f,0.f,0.f,0.f,0.f,0.f,0.f,0.f};
  return z;
}

static __device__ __forceinline__ v8f wmma_bf16(v16bf a, v16bf b, v8f c) {
  // v_wmma_f32_16x16x32_bf16: (neg_a, A, neg_b, B, c_mod, C, reuse_a, reuse_b)
  return __builtin_amdgcn_wmma_f32_16x16x32_bf16(false, a, false, b, (short)0, c,
                                                 false, false);
}

// ---- WMMA fragment loaders (ISA 7.12.2 layouts, wave32) -------------------
// A fragment (16x32, MxK), source row-major [M, ldk]:
//   lane L: row M = L%16; half = L/16 selects K in {h*8..h*8+7} U {16+h*8..16+h*8+7}
static __device__ __forceinline__ v16bf load_a_bf16(const __bf16* row, int k0, int half) {
  v8bf lo = *(const v8bf*)(row + k0 + half * 8);
  v8bf hi = *(const v8bf*)(row + k0 + 16 + half * 8);
  v16bf a;
#pragma unroll
  for (int i = 0; i < 8; ++i) { a[i] = lo[i]; a[i + 8] = hi[i]; }
  return a;
}

// B fragment (32x16, KxN), source is B^T row-major [N, ldk]:
//   lane L: col N = L%16; half = L/16 selects K in {h*8..h*8+7} U {16+h*8..16+h*8+7}
static __device__ __forceinline__ v16bf load_b_bf16(const __bf16* rowN, int k0, int half) {
  v8bf lo = *(const v8bf*)(rowN + k0 + half * 8);
  v8bf hi = *(const v8bf*)(rowN + k0 + 16 + half * 8);
  v16bf b;
#pragma unroll
  for (int i = 0; i < 8; ++i) { b[i] = lo[i]; b[i + 8] = hi[i]; }
  return b;
}

// ---- Stage 0a: zero the BN statistics accumulator -------------------------
__global__ void zero_stats_kernel(float* __restrict__ stats) {
  stats[threadIdx.x] = 0.f;
}

// ---- Stage 0b: one-time f32 -> bf16 downconvert (4 elems/thread) ----------
__global__ __launch_bounds__(256) void cvt_bf16_kernel(
    const float* __restrict__ src, __bf16* __restrict__ dst) {
  const size_t i4 = ((size_t)blockIdx.x * blockDim.x + threadIdx.x) * 4;
  const float4 v = *(const float4*)(src + i4);
  v4bf o;
  o[0] = (__bf16)v.x; o[1] = (__bf16)v.y; o[2] = (__bf16)v.z; o[3] = (__bf16)v.w;
  *(v4bf*)(dst + i4) = o;
}

// ---- Stage 1: Q/K/V per-head projections (bf16 WMMA GEMMs) ----------------
// grid = (SQ/128, B*H, 3); block = 256 (8 waves); wave does 16 rows x 64 cols
// qh[b,h,s,d] (pre-scaled by 1/sqrt(DK)), kh[b,h,t,d] row-major,
// vt[b,h,e,t] TRANSPOSED (so P@V B-fragments + stores are contiguous).
__global__ __launch_bounds__(256) void proj_qkv_kernel(
    const __bf16* __restrict__ qb, const __bf16* __restrict__ xb,
    const __bf16* __restrict__ Wqb, const __bf16* __restrict__ Wkb,
    const __bf16* __restrict__ Wvb,
    __bf16* __restrict__ qh, __bf16* __restrict__ kh, __bf16* __restrict__ vt)
{
  const int lane = threadIdx.x & 31;
  const int wave = threadIdx.x >> 5;
  const int half = lane >> 4;
  const int r    = lane & 15;
  const int mode = blockIdx.z;       // 0: Q, 1: K, 2: V
  const int bh   = blockIdx.y;       // b*H + h
  const int b    = bh >> 3;
  const int h    = bh & 7;
  const int s0   = blockIdx.x * 128 + wave * 16;

  const __bf16* src = (mode == 0) ? (qb + (size_t)b * SQ_ * C_)
                                  : (xb + (size_t)b * SE_ * C_);
  const __bf16* W =
      ((mode == 0) ? Wqb : (mode == 1) ? Wkb : Wvb) + (size_t)h * DK_ * C_;
  const __bf16* arow = src + (size_t)(s0 + r) * C_;

  v8f acc[4];
#pragma unroll
  for (int n = 0; n < 4; ++n) acc[n] = vzero8();

#pragma unroll
  for (int k0 = 0; k0 < C_; k0 += 32) {
    v16bf a = load_a_bf16(arow, k0, half);
#pragma unroll
    for (int n = 0; n < 4; ++n) {
      v16bf bf = load_b_bf16(W + (size_t)(16 * n + r) * C_, k0, half);
      acc[n] = wmma_bf16(a, bf, acc[n]);
    }
  }

  if (mode == 2) {
    // V transposed: vt[bh][e][t]; lane holds 8 consecutive t -> one 16B store
    __bf16* base = vt + (size_t)bh * DV_ * SE_;
#pragma unroll
    for (int n = 0; n < 4; ++n) {
      const int e = 16 * n + r;
      v8bf pk;
#pragma unroll
      for (int i = 0; i < 8; ++i) pk[i] = (__bf16)acc[n][i];
      *(v8bf*)(base + (size_t)e * SE_ + s0 + 8 * half) = pk;
    }
  } else {
    __bf16* base = ((mode == 0) ? qh : kh) + ((size_t)bh * SQ_ + s0) * DK_;
    const float scl = (mode == 0) ? 0.125f : 1.0f;  // fold 1/sqrt(DK) into Q
#pragma unroll
    for (int n = 0; n < 4; ++n)
#pragma unroll
      for (int i = 0; i < 8; ++i)
        base[(size_t)(i + 8 * half) * DK_ + 16 * n + r] = (__bf16)(acc[n][i] * scl);
  }
}

// ---- Stage 2: flash attention (no materialized scores) --------------------
// grid = (SQ/128, B*H); block = 256 (8 waves); wave owns 16 query rows.
// K/V tiles (16 KB each) staged once per block into LDS and shared by all
// 8 waves (cuts K/V read traffic 8x). Per 128-wide key tile:
//   S = Qf @ K^T (WMMA from LDS B-frags) -> online softmax in C-layout regs
//   -> P staged to per-wave LDS (bf16) -> O += P @ V (WMMA from LDS).
__global__ __launch_bounds__(256) void flash_attn_kernel(
    const __bf16* __restrict__ qh, const __bf16* __restrict__ kh,
    const __bf16* __restrict__ vt, __bf16* __restrict__ o)
{
  __shared__ __bf16 Kt[128][64];       // 16 KB  [t_local][d]
  __shared__ __bf16 Vt[64][128];       // 16 KB  [e][t_local]
  __shared__ __bf16 Pl[8][16][128];    // 32 KB  per-wave private P tile
  const int tid  = threadIdx.x;
  const int lane = tid & 31;
  const int wave = tid >> 5;
  const int half = lane >> 4;
  const int r    = lane & 15;
  const int bh   = blockIdx.y;
  const int b    = bh >> 3;
  const int h    = bh & 7;
  const int s0   = blockIdx.x * 128 + wave * 16;

  const __bf16* qrow = qh + ((size_t)bh * SQ_ + s0 + r) * DK_;
  const v16bf Qf0 = load_a_bf16(qrow, 0, half);
  const v16bf Qf1 = load_a_bf16(qrow, 32, half);

  float rm[8], rl[8];
  v8f Oacc[4];
#pragma unroll
  for (int i = 0; i < 8; ++i) { rm[i] = -3.0e38f; rl[i] = 0.f; }
#pragma unroll
  for (int n = 0; n < 4; ++n) Oacc[n] = vzero8();

  for (int t0 = 0; t0 < SE_; t0 += 128) {
    // ---- cooperative staging of K/V tiles (b128 in, b128 to LDS) ----
    {
      // K tile: rows t0..t0+127 x 64 -> one contiguous 16 KB global region
      const uint4* gk = (const uint4*)(kh + ((size_t)bh * SE_ + t0) * DK_);
      uint4* sk = (uint4*)&Kt[0][0];
#pragma unroll
      for (int j = 0; j < 4; ++j) sk[tid + 256 * j] = gk[tid + 256 * j];
      // V^T tile: 64 rows of 128 bf16 (16 uint4 per row), row stride SE_
      uint4* sv = (uint4*)&Vt[0][0];
#pragma unroll
      for (int j = 0; j < 4; ++j) {
        const int idx = tid + 256 * j;   // 0..1023
        const int e   = idx >> 4;
        const int col = idx & 15;
        sv[idx] =
            *((const uint4*)(vt + ((size_t)bh * DV_ + e) * SE_ + t0) + col);
      }
    }
    __syncthreads();

    // ---- scores S = Q @ K^T (B fragments from LDS) ----
    v8f S[8];
#pragma unroll
    for (int n = 0; n < 8; ++n) S[n] = vzero8();
#pragma unroll
    for (int n = 0; n < 8; ++n) {
      const __bf16* krow = &Kt[16 * n + r][0];
      S[n] = wmma_bf16(Qf0, load_b_bf16(krow, 0, half), S[n]);
      S[n] = wmma_bf16(Qf1, load_b_bf16(krow, 32, half), S[n]);
    }

    // ---- online softmax; element i of acc = row (i + 8*half) ----
#pragma unroll
    for (int i = 0; i < 8; ++i) {
      float mx = rm[i];
#pragma unroll
      for (int n = 0; n < 8; ++n) mx = fmaxf(mx, S[n][i]);
#pragma unroll
      for (int off = 1; off < 16; off <<= 1)
        mx = fmaxf(mx, __shfl_xor(mx, off, 32));      // stays within half-wave
      const float alpha = __expf(rm[i] - mx);
      float sum = 0.f;
#pragma unroll
      for (int n = 0; n < 8; ++n) {
        const float pv = __expf(S[n][i] - mx);
        S[n][i] = pv;
        sum += pv;
      }
#pragma unroll
      for (int off = 1; off < 16; off <<= 1)
        sum += __shfl_xor(sum, off, 32);
      rl[i] = rl[i] * alpha + sum;
      rm[i] = mx;
#pragma unroll
      for (int n = 0; n < 4; ++n) Oacc[n][i] *= alpha;
    }

    // ---- stage P (C-layout) -> LDS, reload in A-fragment layout ----
#pragma unroll
    for (int n = 0; n < 8; ++n)
#pragma unroll
      for (int i = 0; i < 8; ++i)
        Pl[wave][i + 8 * half][16 * n + r] = (__bf16)S[n][i];
    __syncthreads();

    // ---- O += P @ V (both operands from LDS) ----
#pragma unroll
    for (int kk = 0; kk < 4; ++kk) {
      v16bf Pf = load_a_bf16(&Pl[wave][r][0], kk * 32, half);
#pragma unroll
      for (int n2 = 0; n2 < 4; ++n2) {
        Oacc[n2] =
            wmma_bf16(Pf, load_b_bf16(&Vt[16 * n2 + r][0], kk * 32, half),
                      Oacc[n2]);
      }
    }
    __syncthreads();   // protect Kt/Vt/Pl before next tile's staging
  }

  // normalize and store head-major concat layout o[b*SQ + s][h*DV + e] (bf16)
  float inv[8];
#pragma unroll
  for (int i = 0; i < 8; ++i) inv[i] = 1.0f / rl[i];
#pragma unroll
  for (int n2 = 0; n2 < 4; ++n2)
#pragma unroll
    for (int i = 0; i < 8; ++i)
      o[((size_t)b * SQ_ + s0 + i + 8 * half) * F_ + h * DV_ + 16 * n2 + r] =
          (__bf16)(Oacc[n2][i] * inv[i]);
}

// ---- Stage 3: output projection p = o @ Wp^T + BN channel statistics ------
// grid = (B*SQ/64); block = 256 (8 waves): 4 row-tiles x 2 column-halves.
__global__ __launch_bounds__(256) void out_proj_kernel(
    const __bf16* __restrict__ o, const __bf16* __restrict__ Wpb,
    float* __restrict__ p, float* __restrict__ stats)
{
  __shared__ float ssum[C_];
  __shared__ float ssq[C_];
  const int tid = threadIdx.x;
  if (tid < C_) { ssum[tid] = 0.f; ssq[tid] = 0.f; }
  __syncthreads();

  const int lane = tid & 31;
  const int wave = tid >> 5;
  const int half = lane >> 4;
  const int r    = lane & 15;
  const int m0 = blockIdx.x * 64 + (wave >> 1) * 16;
  const int c0 = (wave & 1) * 128;

  const __bf16* arow = o + (size_t)(m0 + r) * F_;
  v8f acc[8];
#pragma unroll
  for (int n = 0; n < 8; ++n) acc[n] = vzero8();

#pragma unroll
  for (int k0 = 0; k0 < F_; k0 += 32) {
    v16bf a = load_a_bf16(arow, k0, half);
#pragma unroll
    for (int n = 0; n < 8; ++n) {
      v16bf bf = load_b_bf16(Wpb + (size_t)(c0 + 16 * n + r) * F_, k0, half);
      acc[n] = wmma_bf16(a, bf, acc[n]);
    }
  }

#pragma unroll
  for (int n = 0; n < 8; ++n) {
    const int c = c0 + 16 * n + r;
    float ls = 0.f, lq = 0.f;
#pragma unroll
    for (int i = 0; i < 8; ++i) {
      const float v = acc[n][i];
      p[(size_t)(m0 + i + 8 * half) * C_ + c] = v;
      ls += v;
      lq += v * v;
    }
    ls += __shfl_xor(ls, 16, 32);   // fold the two half-waves (same channel)
    lq += __shfl_xor(lq, 16, 32);
    if (half == 0) {
      atomicAdd(&ssum[c], ls);       // ds_add_f32
      atomicAdd(&ssq[c], lq);
    }
  }
  __syncthreads();
  if (tid < C_) {
    atomicAdd(&stats[tid], ssum[tid]);
    atomicAdd(&stats[C_ + tid], ssq[tid]);
  }
}

// ---- Stage 4: BatchNorm (batch stats) + LeakyReLU, float4 vectorized ------
__global__ __launch_bounds__(256) void bn_act_kernel(
    const float* __restrict__ p, const float* __restrict__ stats,
    const float* __restrict__ gamma, const float* __restrict__ beta,
    float* __restrict__ y)
{
  const size_t idx4 = ((size_t)blockIdx.x * blockDim.x + threadIdx.x) * 4;
  const int c = (int)(idx4 & (C_ - 1));
  const float rcnt = 1.0f / (float)(B_ * SQ_);
  float pv[4], yv[4];
  *(float4*)pv = *(const float4*)(p + idx4);
#pragma unroll
  for (int j = 0; j < 4; ++j) {
    const int cj = c + j;
    const float mean = stats[cj] * rcnt;
    const float var  = stats[C_ + cj] * rcnt - mean * mean;
    const float t = (pv[j] - mean) * rsqrtf(var + BN_EPS) * gamma[cj] + beta[cj];
    yv[j] = (t >= 0.f) ? t : NEG_SLOPE * t;
  }
  *(float4*)(y + idx4) = *(float4*)yv;
}

extern "C" void kernel_launch(void* const* d_in, const int* in_sizes, int n_in,
                              void* d_out, int out_size, void* d_ws, size_t ws_size,
                              hipStream_t stream) {
  const float* x     = (const float*)d_in[0];
  const float* q     = (const float*)d_in[1];
  const float* Wq    = (const float*)d_in[2];
  const float* Wk    = (const float*)d_in[3];
  const float* Wv    = (const float*)d_in[4];
  const float* Wp    = (const float*)d_in[5];
  const float* gamma = (const float*)d_in[6];
  const float* beta  = (const float*)d_in[7];
  float* y = (float*)d_out;

  char* ws = (char*)d_ws;
  __bf16* qh   = (__bf16*)(ws + (0ull  << 20));   // 16 MB [B,H,SQ,DK] bf16
  __bf16* kh   = (__bf16*)(ws + (16ull << 20));   // 16 MB [B,H,SE,DK] bf16
  __bf16* vt   = (__bf16*)(ws + (32ull << 20));   // 16 MB [B,H,DV,SE] bf16 (V^T)
  __bf16* o    = (__bf16*)(ws + (48ull << 20));   // 16 MB [B*SQ, F]   bf16
  float*  p    = (float*) (ws + (64ull << 20));   // 16 MB [B*SQ, C]   f32
  float*  stat = (float*) (ws + (80ull << 20));   // 2*C floats: sum, sumsq
  __bf16* qb   = (__bf16*)(ws + (81ull << 20));   // 8 MB  q in bf16
  __bf16* xb   = (__bf16*)(ws + (89ull << 20));   // 8 MB  x in bf16
  __bf16* Wqb  = (__bf16*)(ws + (97ull << 20));   // 256 KB
  __bf16* Wkb  = (__bf16*)(ws + (98ull << 20));   // 256 KB
  __bf16* Wvb  = (__bf16*)(ws + (99ull << 20));   // 256 KB
  __bf16* Wpb  = (__bf16*)(ws + (100ull << 20));  // 256 KB

  zero_stats_kernel<<<1, 2 * C_, 0, stream>>>(stat);
  // one-time bf16 downconversions (1024 f32 elems per block)
  cvt_bf16_kernel<<<(B_ * SQ_ * C_) / 1024, 256, 0, stream>>>(q, qb);
  cvt_bf16_kernel<<<(B_ * SE_ * C_) / 1024, 256, 0, stream>>>(x, xb);
  cvt_bf16_kernel<<<(H_ * DK_ * C_) / 1024, 256, 0, stream>>>(Wq, Wqb);
  cvt_bf16_kernel<<<(H_ * DK_ * C_) / 1024, 256, 0, stream>>>(Wk, Wkb);
  cvt_bf16_kernel<<<(H_ * DV_ * C_) / 1024, 256, 0, stream>>>(Wv, Wvb);
  cvt_bf16_kernel<<<(C_ * F_) / 1024, 256, 0, stream>>>(Wp, Wpb);

  proj_qkv_kernel<<<dim3(SQ_ / 128, B_ * H_, 3), 256, 0, stream>>>(
      qb, xb, Wqb, Wkb, Wvb, qh, kh, vt);
  flash_attn_kernel<<<dim3(SQ_ / 128, B_ * H_), 256, 0, stream>>>(qh, kh, vt, o);
  out_proj_kernel<<<dim3((B_ * SQ_) / 64), 256, 0, stream>>>(o, Wpb, p, stat);
  bn_act_kernel<<<dim3((B_ * SQ_ * C_) / (256 * 4)), 256, 0, stream>>>(
      p, stat, gamma, beta, y);
}
